// AttentionBlock_56341380989380
// MI455X (gfx1250) — matrix operand order repaired and checked by reading, not compiled, not run on hardware
//
#include <hip/hip_runtime.h>
#include <hip/hip_bf16.h>
#include <math.h>

typedef __bf16 bf16_t;
typedef __attribute__((ext_vector_type(16))) __bf16 v16bf;
typedef __attribute__((ext_vector_type(8)))  float   v8f;
typedef __attribute__((ext_vector_type(4)))  unsigned int v4u;

#define S_LEN 2048
#define HDIM  2048
#define NHEAD 16
#define HEADD 128
#define INTER_DIM 8192
#define MASK_VAL -10000000000.0f

// ---------------------------------------------------------------------------
// CDNA5 async memory->LDS copy (ASYNCcnt-tracked), 16 bytes per lane.
// vdst = raw LDS byte address (AS3 pointer), vaddr = 64-bit global address.
// ---------------------------------------------------------------------------
typedef __attribute__((address_space(3))) bf16_t* lds_bf16_ptr;
typedef const __attribute__((address_space(1))) bf16_t* gbl_bf16_ptr;

__device__ __forceinline__ void async_copy_b128(bf16_t* lds_dst, const bf16_t* gsrc) {
  lds_bf16_ptr l = (lds_bf16_ptr)lds_dst;   // addrspacecast generic -> LDS
  gbl_bf16_ptr g = (gbl_bf16_ptr)gsrc;      // addrspacecast generic -> global
  asm volatile("global_load_async_to_lds_b128 %0, %1, off"
               :
               : "v"(l), "v"(g)
               : "memory");
}

#define WAIT_ASYNC_LE(imm) asm volatile("s_wait_asynccnt " #imm ::: "memory")

// ---------------------------------------------------------------------------
// fp32 -> bf16 cast (grid-stride)
// ---------------------------------------------------------------------------
__global__ __launch_bounds__(256) void cast_f32_bf16(const float* __restrict__ src,
                                                     bf16_t* __restrict__ dst, size_t n) {
  size_t i = (size_t)blockIdx.x * 256 + threadIdx.x;
  size_t stride = (size_t)gridDim.x * 256;
  for (; i < n; i += stride) dst[i] = (bf16_t)src[i];
}

// ---------------------------------------------------------------------------
// Tiled transpose + fp32->bf16 cast: src [Kr, Nc] row-major -> dst [Nc, Kr].
// 64x64 tile per block, coalesced on both the read and write side.
// ---------------------------------------------------------------------------
__global__ __launch_bounds__(256) void transpose_cast_f32_bf16(
    const float* __restrict__ src, bf16_t* __restrict__ dst, int Kr, int Nc) {
  __shared__ bf16_t t[64][65];
  const int kb = blockIdx.y * 64;
  const int nb = blockIdx.x * 64;
  const int tid = threadIdx.x;
  for (int i = tid; i < 4096; i += 256) {
    int k = i >> 6, n = i & 63;  // consecutive tid -> consecutive n (coalesced f32 read)
    t[n][k] = (bf16_t)src[(size_t)(kb + k) * Nc + (nb + n)];
  }
  __syncthreads();
  for (int i = tid; i < 4096; i += 256) {
    int n = i >> 6, k = i & 63;  // consecutive tid -> consecutive k (coalesced bf16 write)
    dst[(size_t)(nb + n) * Kr + (kb + k)] = t[n][k];
  }
}

// ---------------------------------------------------------------------------
// LayerNorm: out_b = LN(xa [+ xb]) * g + b  (bf16), optional resid = xa+xb
// ---------------------------------------------------------------------------
__global__ __launch_bounds__(256) void layernorm_kernel(
    const float* __restrict__ xa, const float* __restrict__ xb,
    const float* __restrict__ g, const float* __restrict__ b,
    float* __restrict__ resid_out, bf16_t* __restrict__ out_b) {
  const int row = blockIdx.x;
  const int tid = threadIdx.x;
  const float* pa = xa + (size_t)row * HDIM;
  const float* pb = xb ? xb + (size_t)row * HDIM : nullptr;

  float vals[8];
  float lsum = 0.f, lsq = 0.f;
#pragma unroll
  for (int i = 0; i < 8; ++i) {
    int c = tid + i * 256;
    float v = pa[c];
    if (pb) v += pb[c];
    vals[i] = v;
    lsum += v; lsq += v * v;
    if (resid_out) resid_out[(size_t)row * HDIM + c] = v;
  }
  __shared__ float s1[256], s2[256];
  s1[tid] = lsum; s2[tid] = lsq;
  __syncthreads();
  for (int s = 128; s > 0; s >>= 1) {
    if (tid < s) { s1[tid] += s1[tid + s]; s2[tid] += s2[tid + s]; }
    __syncthreads();
  }
  const float mean = s1[0] * (1.0f / HDIM);
  const float var  = s2[0] * (1.0f / HDIM) - mean * mean;
  const float rs   = rsqrtf(var + 1e-5f);
#pragma unroll
  for (int i = 0; i < 8; ++i) {
    int c = tid + i * 256;
    out_b[(size_t)row * HDIM + c] = (bf16_t)((vals[i] - mean) * rs * g[c] + b[c]);
  }
}

// ---------------------------------------------------------------------------
// RoPE (GPT-J interleaved, full rotary) + 1/sqrt(HD) on q; repack fp32 [S,H]
// -> bf16 head-major [NH][S][HD] for q and k (V handled by full transpose).
// ---------------------------------------------------------------------------
__global__ __launch_bounds__(256) void rope_repack_kernel(
    const float* __restrict__ qf, const float* __restrict__ kf,
    const int* __restrict__ pos_ids,
    bf16_t* __restrict__ qb, bf16_t* __restrict__ kb) {
  size_t idx = (size_t)blockIdx.x * 256 + threadIdx.x;
  const size_t total = (size_t)S_LEN * NHEAD * (HEADD / 2);
  if (idx >= total) return;
  const int d2 = (int)(idx % (HEADD / 2));
  const int h  = (int)((idx / (HEADD / 2)) % NHEAD);
  const int s  = (int)(idx / ((size_t)(HEADD / 2) * NHEAD));
  const int d0 = d2 * 2;

  const size_t src = (size_t)s * HDIM + h * HEADD + d0;
  const float q0 = qf[src], q1 = qf[src + 1];
  const float k0 = kf[src], k1 = kf[src + 1];

  const float p    = (float)pos_ids[s];
  const float freq = powf(10000.0f, -(float)d0 / (float)HEADD);
  const float ang  = p * freq;
  const float sn = sinf(ang), cs = cosf(ang);
  const float qs = 0.08838834764831845f;  // 1/sqrt(128)

  const size_t dst = ((size_t)h * S_LEN + s) * HEADD + d0;
  qb[dst]     = (bf16_t)((q0 * cs - q1 * sn) * qs);
  qb[dst + 1] = (bf16_t)((q1 * cs + q0 * sn) * qs);
  kb[dst]     = (bf16_t)(k0 * cs - k1 * sn);
  kb[dst + 1] = (bf16_t)(k1 * cs + k0 * sn);
}

// ---------------------------------------------------------------------------
// Row softmax over fp32 scores -> bf16 probabilities. One block per row.
// ---------------------------------------------------------------------------
__global__ __launch_bounds__(256) void softmax_kernel(const float* __restrict__ sc,
                                                      bf16_t* __restrict__ out) {
  const int row = blockIdx.x;
  const int tid = threadIdx.x;
  const float* p = sc + (size_t)row * S_LEN;
  __shared__ float red[256];

  float m = -3.0e38f;
  for (int c = tid; c < S_LEN; c += 256) m = fmaxf(m, p[c]);
  red[tid] = m; __syncthreads();
  for (int s = 128; s > 0; s >>= 1) {
    if (tid < s) red[tid] = fmaxf(red[tid], red[tid + s]);
    __syncthreads();
  }
  const float rmax = red[0];
  __syncthreads();

  float sum = 0.f;
  for (int c = tid; c < S_LEN; c += 256) sum += __expf(p[c] - rmax);
  red[tid] = sum; __syncthreads();
  for (int s = 128; s > 0; s >>= 1) {
    if (tid < s) red[tid] += red[tid + s];
    __syncthreads();
  }
  const float inv = 1.0f / red[0];
  for (int c = tid; c < S_LEN; c += 256)
    out[(size_t)row * S_LEN + c] = (bf16_t)(__expf(p[c] - rmax) * inv);
}

// ---------------------------------------------------------------------------
// bf16 WMMA GEMM: C[M,N] = A[M,K] @ B^T + bias + addsrc (+GELU / causal mask).
// A row-major [M,K], B row-major [N,K] (all operands pre-transposed so every
// tile stage is contiguous 16B chunks -> async-to-LDS).
// 128x128 C-tile per 256-thread block (8 waves, 32x64 per wave, 8 WMMA accums)
// BK=32, double-buffered LDS, staged entirely with
// global_load_async_to_lds_b128 + s_wait_asynccnt (CDNA5 async path).
// ---------------------------------------------------------------------------
__device__ __forceinline__ void stage_tile_async(
    const bf16_t* __restrict__ A, int lda, const bf16_t* __restrict__ B, int ldb,
    int m0, int n0, int k0, bf16_t* sA, bf16_t* sB, int tid) {
#pragma unroll
  for (int j = 0; j < 2; ++j) {
    int i = tid + j * 256;          // 512 chunks of 8 bf16 for the A tile
    int m = i >> 2;
    int kc = (i & 3) << 3;
    async_copy_b128(&sA[m * 40 + kc], &A[(size_t)(m0 + m) * lda + (k0 + kc)]);
  }
#pragma unroll
  for (int j = 0; j < 2; ++j) {
    int i = tid + j * 256;          // 512 chunks of 8 bf16 for the B tile
    int n = i >> 2;
    int kc = (i & 3) << 3;
    async_copy_b128(&sB[n * 40 + kc], &B[(size_t)(n0 + n) * ldb + (k0 + kc)]);
  }
}

template <bool CAUSAL, bool GELU_ACT, bool OUT_BF16>
__global__ __launch_bounds__(256) void gemm_wmma(
    const bf16_t* __restrict__ A, int lda,
    const bf16_t* __restrict__ B, int ldb,       // [N,K] row-major
    const float* __restrict__ bias,              // [N] or null
    const float* __restrict__ addsrc,            // [M,ldc] or null
    const float* __restrict__ scalar_add,        // 1 element or null
    float* __restrict__ Cf, bf16_t* __restrict__ Cb,
    int ldc, int M, int N, int K) {
  const int m0 = blockIdx.y * 128;
  const int n0 = blockIdx.x * 128;
  const int tid = threadIdx.x;

  if (CAUSAL && !OUT_BF16) {
    if (n0 >= m0 + 128) {  // tile entirely above the diagonal: mask, no compute
      for (int i = tid; i < 128 * 128; i += 256) {
        int r = i >> 7, c = i & 127;
        Cf[(size_t)(m0 + r) * ldc + (n0 + c)] = MASK_VAL;
      }
      return;
    }
  }

  __shared__ __align__(16) bf16_t sA[2][128 * 40];
  __shared__ __align__(16) bf16_t sB[2][128 * 40];

  const int wave = tid >> 5;
  const int lane = tid & 31;
  const int wm = (wave >> 1) * 32;  // 0,32,64,96
  const int wn = (wave & 1) * 64;   // 0,64
  const int half = lane >> 4;
  const int l16 = lane & 15;

  v8f acc[2][4];
#pragma unroll
  for (int im = 0; im < 2; ++im)
#pragma unroll
    for (int in = 0; in < 4; ++in)
      acc[im][in] = (v8f){0.f, 0.f, 0.f, 0.f, 0.f, 0.f, 0.f, 0.f};

  const int nk = K >> 5;
  stage_tile_async(A, lda, B, ldb, m0, n0, 0, sA[0], sB[0], tid);

  for (int it = 0; it < nk; ++it) {
    const int cur = it & 1;
    if (it + 1 < nk) {
      // Prefetch next K-tile into the other buffer, keep its 4 ops in flight,
      // but require the current tile's 4 ops to have landed.
      stage_tile_async(A, lda, B, ldb, m0, n0, (it + 1) << 5,
                       sA[cur ^ 1], sB[cur ^ 1], tid);
      WAIT_ASYNC_LE(0x4);
    } else {
      WAIT_ASYNC_LE(0x0);
    }
    __syncthreads();

    const bf16_t* cA = sA[cur];
    const bf16_t* cB = sB[cur];

    // A fragments: lane half 0 -> K{0..7,16..23}, half 1 -> K{8..15,24..31}.
    v16bf afrag[2];
#pragma unroll
    for (int im = 0; im < 2; ++im) {
      const bf16_t* base = &cA[(wm + im * 16 + l16) * 40];
      union { v16bf v; v4u q[2]; } u;
      u.q[0] = *(const v4u*)(base + half * 8);
      u.q[1] = *(const v4u*)(base + 16 + half * 8);
      afrag[im] = u.v;
    }
    // B fragments: lane half 0 -> K{0..7,16..23}, half 1 -> K{8..15,24..31}.
    v16bf bfrag[4];
#pragma unroll
    for (int in = 0; in < 4; ++in) {
      const bf16_t* base = &cB[(wn + in * 16 + l16) * 40] + half * 8;
      union { v16bf v; v4u q[2]; } u;
      u.q[0] = *(const v4u*)(base);
      u.q[1] = *(const v4u*)(base + 16);
      bfrag[in] = u.v;
    }
#pragma unroll
    for (int im = 0; im < 2; ++im)
#pragma unroll
      for (int in = 0; in < 4; ++in)
        acc[im][in] = __builtin_amdgcn_wmma_f32_16x16x32_bf16(
            false, afrag[im], false, bfrag[in], (short)0, acc[im][in],
            false, false);
    __syncthreads();  // all waves done reading this buffer before it is refilled
  }

  // Epilogue: C element (m,n) lives in vgpr p = m%8, lane = (m/8)*16 + n.
#pragma unroll
  for (int im = 0; im < 2; ++im) {
#pragma unroll
    for (int in = 0; in < 4; ++in) {
      const int grow_base = m0 + wm + im * 16 + half * 8;
      const int gcol = n0 + wn + in * 16 + l16;
      const float bv = bias ? bias[gcol] : 0.f;
#pragma unroll
      for (int p = 0; p < 8; ++p) {
        const int grow = grow_base + p;
        float v = acc[im][in][p] + bv;
        const size_t off = (size_t)grow * ldc + gcol;
        if (addsrc) v += addsrc[off];
        if (scalar_add) v += scalar_add[0];
        if (CAUSAL && (grow < gcol)) v = MASK_VAL;
        if (GELU_ACT) v = 0.5f * v * (1.0f + erff(v * 0.70710678118654752f));
        if (OUT_BF16) Cb[off] = (bf16_t)v;
        else          Cf[off] = v;
      }
    }
  }
}

// ---------------------------------------------------------------------------
// Host-side orchestration.
// ---------------------------------------------------------------------------
extern "C" void kernel_launch(void* const* d_in, const int* in_sizes, int n_in,
                              void* d_out, int out_size, void* d_ws, size_t ws_size,
                              hipStream_t stream) {
  const float* x      = (const float*)d_in[0];
  const float* abias  = (const float*)d_in[1];
  const float* wq     = (const float*)d_in[2];
  const float* wk     = (const float*)d_in[3];
  const float* wv     = (const float*)d_in[4];
  const float* wo     = (const float*)d_in[5];
  const float* ln1_g  = (const float*)d_in[6];
  const float* ln1_b  = (const float*)d_in[7];
  const float* ln2_g  = (const float*)d_in[8];
  const float* ln2_b  = (const float*)d_in[9];
  const float* w_in   = (const float*)d_in[10];
  const float* b_in   = (const float*)d_in[11];
  const float* w_out  = (const float*)d_in[12];
  const float* b_out  = (const float*)d_in[13];
  const int*   pos    = (const int*)d_in[14];
  float* out = (float*)d_out;
  char* ws = (char*)d_ws;

  const size_t MB = 1024u * 1024u;
  bf16_t* HB    = (bf16_t*)(ws + 0 * MB);     // 8 MB  LN1 output [S,H]
  bf16_t* WQT   = (bf16_t*)(ws + 8 * MB);     // 8 MB  wq^T [N,K]
  bf16_t* WKT   = (bf16_t*)(ws + 16 * MB);    // 8 MB
  bf16_t* WVT   = (bf16_t*)(ws + 24 * MB);    // 8 MB
  bf16_t* WOT   = (bf16_t*)(ws + 32 * MB);    // 8 MB
  bf16_t* WINT  = (bf16_t*)(ws + 40 * MB);    // 32 MB w_in^T  [INTER,H]
  bf16_t* WOUTT = (bf16_t*)(ws + 72 * MB);    // 32 MB w_out^T [H,INTER]
  float*  QF    = (float*)(ws + 104 * MB);    // 16 MB (later: CTXB alias)
  float*  KF    = (float*)(ws + 120 * MB);    // 16 MB (later: ATTNOUT alias)
  float*  VF    = (float*)(ws + 136 * MB);    // 16 MB (later: RESID2 alias)
  bf16_t* QB    = (bf16_t*)(ws + 152 * MB);   // 8 MB  q head-major [NH][S][HD]
  bf16_t* KB_   = (bf16_t*)(ws + 160 * MB);   // 8 MB  k head-major [NH][S][HD]
  bf16_t* VBT   = (bf16_t*)(ws + 168 * MB);   // 8 MB  v^T [H,S] == [NH][HD][S]
  float*  SC    = (float*)(ws + 176 * MB);    // 16 MB per-head scores
  bf16_t* ATT   = (bf16_t*)(ws + 192 * MB);   // 8 MB  per-head probs
  bf16_t* H2B   = (bf16_t*)(ws + 200 * MB);   // 8 MB  LN2 output
  bf16_t* MIDB  = (bf16_t*)(ws + 208 * MB);   // 32 MB FFN hidden (ends 240 MB)
  bf16_t* CTXB    = (bf16_t*)QF;  // 8 MB, reuses QF (dead after rope)
  float*  ATTNOUT = KF;           // reuses KF
  float*  RESID2  = VF;           // reuses VF

  const dim3 blk(256);

  // 1) LN1 -> bf16 activations
  layernorm_kernel<<<S_LEN, blk, 0, stream>>>(x, nullptr, ln1_g, ln1_b, nullptr, HB);

  // 2) Weight transpose+cast fp32 -> bf16 [N,K]
  {
    dim3 g_sq(HDIM / 64, HDIM / 64);
    transpose_cast_f32_bf16<<<g_sq, blk, 0, stream>>>(wq, WQT, HDIM, HDIM);
    transpose_cast_f32_bf16<<<g_sq, blk, 0, stream>>>(wk, WKT, HDIM, HDIM);
    transpose_cast_f32_bf16<<<g_sq, blk, 0, stream>>>(wv, WVT, HDIM, HDIM);
    transpose_cast_f32_bf16<<<g_sq, blk, 0, stream>>>(wo, WOT, HDIM, HDIM);
    dim3 g_in(INTER_DIM / 64, HDIM / 64);   // src [H, INTER]
    transpose_cast_f32_bf16<<<g_in, blk, 0, stream>>>(w_in, WINT, HDIM, INTER_DIM);
    dim3 g_out(HDIM / 64, INTER_DIM / 64);  // src [INTER, H]
    transpose_cast_f32_bf16<<<g_out, blk, 0, stream>>>(w_out, WOUTT, INTER_DIM, HDIM);
  }

  // 3) QKV projections: h @ Wq/Wk/Wv (B supplied as W^T)
  {
    dim3 grid(HDIM / 128, S_LEN / 128);
    gemm_wmma<false, false, false><<<grid, blk, 0, stream>>>(
        HB, HDIM, WQT, HDIM, nullptr, nullptr, nullptr, QF, nullptr,
        HDIM, S_LEN, HDIM, HDIM);
    gemm_wmma<false, false, false><<<grid, blk, 0, stream>>>(
        HB, HDIM, WKT, HDIM, nullptr, nullptr, nullptr, KF, nullptr,
        HDIM, S_LEN, HDIM, HDIM);
    gemm_wmma<false, false, false><<<grid, blk, 0, stream>>>(
        HB, HDIM, WVT, HDIM, nullptr, nullptr, nullptr, VF, nullptr,
        HDIM, S_LEN, HDIM, HDIM);
  }

  // 4) RoPE + scale + repack q,k head-major; v via full transpose [S,H]->[H,S]
  rope_repack_kernel<<<(S_LEN * NHEAD * (HEADD / 2)) / 256, blk, 0, stream>>>(
      QF, KF, pos, QB, KB_);
  {
    dim3 g_v(HDIM / 64, S_LEN / 64);
    transpose_cast_f32_bf16<<<g_v, blk, 0, stream>>>(VF, VBT, S_LEN, HDIM);
  }

  // 5) Attention per head: scores (q @ k^T, causal) -> softmax -> ctx
  for (int h = 0; h < NHEAD; ++h) {
    const bf16_t* qh  = QB  + (size_t)h * S_LEN * HEADD;
    const bf16_t* kh  = KB_ + (size_t)h * S_LEN * HEADD;
    const bf16_t* vht = VBT + (size_t)h * HEADD * S_LEN;  // [HD, S] = [N,K]
    {
      dim3 grid(S_LEN / 128, S_LEN / 128);
      gemm_wmma<true, false, false><<<grid, blk, 0, stream>>>(
          qh, HEADD, kh, HEADD, nullptr, nullptr, abias, SC, nullptr,
          S_LEN, S_LEN, S_LEN, HEADD);
    }
    softmax_kernel<<<S_LEN, blk, 0, stream>>>(SC, ATT);
    {
      dim3 grid(HEADD / 128, S_LEN / 128);  // (1,16)
      gemm_wmma<false, false, true><<<grid, blk, 0, stream>>>(
          ATT, S_LEN, vht, S_LEN, nullptr, nullptr, nullptr, nullptr,
          CTXB + (size_t)h * HEADD, HDIM, S_LEN, HEADD, S_LEN);
    }
  }

  // 6) Output projection: ctx @ wo -> attn_out (fp32)
  {
    dim3 grid(HDIM / 128, S_LEN / 128);
    gemm_wmma<false, false, false><<<grid, blk, 0, stream>>>(
        CTXB, HDIM, WOT, HDIM, nullptr, nullptr, nullptr, ATTNOUT, nullptr,
        HDIM, S_LEN, HDIM, HDIM);
  }

  // 7) LN2 over (attn_out + x), keep the residual sum
  layernorm_kernel<<<S_LEN, blk, 0, stream>>>(ATTNOUT, x, ln2_g, ln2_b, RESID2, H2B);

  // 8) FFN up: + b_in, exact GELU, bf16 out
  {
    dim3 grid(INTER_DIM / 128, S_LEN / 128);
    gemm_wmma<false, true, true><<<grid, blk, 0, stream>>>(
        H2B, HDIM, WINT, HDIM, b_in, nullptr, nullptr, nullptr, MIDB,
        INTER_DIM, S_LEN, INTER_DIM, HDIM);
  }

  // 9) FFN down: + b_out + (attn_out + x) -> final output (fp32)
  {
    dim3 grid(HDIM / 128, S_LEN / 128);
    gemm_wmma<false, false, false><<<grid, blk, 0, stream>>>(
        MIDB, INTER_DIM, WOUTT, INTER_DIM, b_out, RESID2, nullptr, out, nullptr,
        HDIM, S_LEN, HDIM, INTER_DIM);
  }
}
